// Decoder_65652870087141
// MI455X (gfx1250) — matrix operand order repaired and checked by reading, not compiled, not run on hardware
//
#include <hip/hip_runtime.h>

// ---------------------------------------------------------------------------
// LAS-style attention decoder for MI455X (gfx1250, wave32).
// Big LSTM GEMMs run on v_wmma_f32_16x16x32_bf16 (bf16 operands, f32 accum);
// weights are converted to bf16 once per launch (~50MB, stays hot in 192MB L2
// across the 100 sequential steps). Gate epilogue (sigmoid/tanh cell update)
// is fused into the WMMA kernel so gates never hit memory.
// Each wave owns TWO 16-row M-tiles so every B (weight) fragment feeds two
// WMMAs: 12 b128 loads per 8 WMMAs (1.5 loads/WMMA) instead of 2.5.
// ---------------------------------------------------------------------------

typedef __attribute__((ext_vector_type(16))) __bf16 v16bf;
typedef __attribute__((ext_vector_type(8)))  __bf16 v8bf;
typedef __attribute__((ext_vector_type(8)))  float  v8f;

static constexpr int Bn = 64;    // batch
static constexpr int Tn = 100;   // time steps
static constexpr int Sn = 400;   // source length
static constexpr int En = 512;   // embedding dim
static constexpr int Hn = 1024;  // hidden
static constexpr int Kn = 512;   // key/value dim
static constexpr int Vn = 34;    // vocab

__device__ __forceinline__ unsigned short f2bf(float x) {
  unsigned u = __float_as_uint(x);
  unsigned r = u + 0x7FFFu + ((u >> 16) & 1u);   // round-to-nearest-even
  return (unsigned short)(r >> 16);
}
__device__ __forceinline__ float sigmoidf(float x) { return 1.f / (1.f + __expf(-x)); }

__device__ __forceinline__ v8f wmma_bf16(v16bf a, v16bf b, v8f c) {
  // D = A(16x32 bf16) * B(32x16 bf16) + C(16x16 f32)
  return __builtin_amdgcn_wmma_f32_16x16x32_bf16(false, a, false, b, (short)0, c,
                                                 false, false);
}

// A-fragment: 16x32 bf16 tile. lane holds row (lane&15); 16 contiguous K
// values at k+(lane>>4)*16 (matches B fragment order).
__device__ __forceinline__ v16bf load_a_frag(const unsigned short* base, int row,
                                             int lda, int k, int laneH) {
  const __bf16* p = (const __bf16*)(base + row * lda + k + laneH * 16);
  v8bf lo = *(const v8bf*)p;
  v8bf hi = *(const v8bf*)(p + 8);
  v16bf a;
#pragma unroll
  for (int i = 0; i < 8; ++i) { a[i] = lo[i]; a[i + 8] = hi[i]; }
  return a;
}

// B-fragment: 32x16 bf16 tile from weight stored [N][K] row-major.
// lane holds col j=(lane&15); 16 contiguous K values at k+(lane>>4)*16.
__device__ __forceinline__ v16bf load_b_frag(const unsigned short* w, int jrow,
                                             int ldw, int k, int laneH) {
  const __bf16* p = (const __bf16*)(w + jrow * ldw + k + laneH * 16);
  return *(const v16bf*)p;
}

// ---------------------------------------------------------------------------
// Fused LSTM cell: gates = [a0|a1] @ w_ih^T + hprev @ w_hh^T + b, then
// c = sig(f)*c + sig(i)*tanh(g); h = sig(o)*tanh(c).
// Wave computes 32(batch) x 16(hidden) for ALL FOUR gates: acc[2][4],
// B fragment loaded once per (gate,k) and reused for both M-tiles.
// Block = 128 thr (4 waves): 2 across M(=64), 2 across 16-col slices.
// Grid = H/32 = 32 blocks.
// ---------------------------------------------------------------------------
__global__ __launch_bounds__(128) void lstm_cell_kernel(
    const unsigned short* __restrict__ a0, int k0, int lda0,
    const unsigned short* __restrict__ a1, int k1, int lda1,
    const unsigned short* __restrict__ hprev,
    const unsigned short* __restrict__ w_ih, int ldw_ih,
    const unsigned short* __restrict__ w_hh,
    const float* __restrict__ b_ih, const float* __restrict__ b_hh,
    float* __restrict__ c_state,                 // in/out fp32 [B,H]
    unsigned short* __restrict__ h_out_bf)       // out bf16 [B,H]
{
  const int tid = threadIdx.x;
  const int lane = tid & 31;
  const int wid = tid >> 5;                            // 0..3
  const int laneN = lane & 15;
  const int laneH = lane >> 4;
  const int m0 = (wid & 1) * 32;                       // batch tile base (2 tiles)
  const int n0 = blockIdx.x * 32 + (wid >> 1) * 16;    // hidden-col tile base
  const int mrow0 = m0 + laneN;
  const int mrow1 = m0 + 16 + laneN;

  v8f zero = {0.f, 0.f, 0.f, 0.f, 0.f, 0.f, 0.f, 0.f};
  v8f acc[2][4];
#pragma unroll
  for (int mt = 0; mt < 2; ++mt)
#pragma unroll
    for (int g = 0; g < 4; ++g) acc[mt][g] = zero;

  auto accum = [&](const unsigned short* A, int lda, const unsigned short* W,
                   int ldw, int wcol, int klen) {
    for (int k = 0; k < klen; k += 32) {
      v16bf af0 = load_a_frag(A, mrow0, lda, k, laneH);
      v16bf af1 = load_a_frag(A, mrow1, lda, k, laneH);
#pragma unroll
      for (int g = 0; g < 4; ++g) {
        v16bf bf = load_b_frag(W, g * Hn + n0 + laneN, ldw, wcol + k, laneH);
        acc[0][g] = wmma_bf16(af0, bf, acc[0][g]);
        acc[1][g] = wmma_bf16(af1, bf, acc[1][g]);
      }
    }
  };

  accum(a0, lda0, w_ih, ldw_ih, 0, k0);
  if (a1) accum(a1, lda1, w_ih, ldw_ih, k0, k1);
  accum(hprev, Hn, w_hh, Hn, 0, Hn);

  // Epilogue: lane holds col n, rows (tile base)+laneH*8+r (r=0..7).
  const int n = n0 + laneN;
  const float bi = b_ih[n] + b_hh[n];
  const float bf_ = b_ih[Hn + n] + b_hh[Hn + n];
  const float bg = b_ih[2 * Hn + n] + b_hh[2 * Hn + n];
  const float bo = b_ih[3 * Hn + n] + b_hh[3 * Hn + n];
#pragma unroll
  for (int mt = 0; mt < 2; ++mt) {
    const int mb = m0 + mt * 16 + laneH * 8;
#pragma unroll
    for (int r = 0; r < 8; ++r) {
      int idx = (mb + r) * Hn + n;
      float iv = sigmoidf(acc[mt][0][r] + bi);
      float fv = sigmoidf(acc[mt][1][r] + bf_);
      float gv = tanhf(acc[mt][2][r] + bg);
      float ov = sigmoidf(acc[mt][3][r] + bo);
      float c = fv * c_state[idx] + iv * gv;
      float h = ov * tanhf(c);
      c_state[idx] = c;
      h_out_bf[idx] = f2bf(h);
    }
  }
}

// ---------------------------------------------------------------------------
// Generic WMMA GEMM + bias: out[M=64,N] = A[64,K]bf16 @ W[N,K]bf16^T + bias.
// Wave owns 2 M-tiles (B fragment reused); block 128 thr; grid = N/32.
// ---------------------------------------------------------------------------
__global__ __launch_bounds__(128) void gemm_bias_kernel(
    const unsigned short* __restrict__ A, int lda, int Kdim,
    const unsigned short* __restrict__ W, const float* __restrict__ bias,
    float* __restrict__ outF, unsigned short* __restrict__ outBF, int N)
{
  const int tid = threadIdx.x;
  const int lane = tid & 31;
  const int wid = tid >> 5;
  const int laneN = lane & 15;
  const int laneH = lane >> 4;
  const int m0 = (wid & 1) * 32;
  const int n0 = blockIdx.x * 32 + (wid >> 1) * 16;
  const int mrow0 = m0 + laneN;
  const int mrow1 = m0 + 16 + laneN;

  v8f zero = {0.f, 0.f, 0.f, 0.f, 0.f, 0.f, 0.f, 0.f};
  v8f acc0 = zero, acc1 = zero;
  for (int k = 0; k < Kdim; k += 32) {
    v16bf af0 = load_a_frag(A, mrow0, lda, k, laneH);
    v16bf af1 = load_a_frag(A, mrow1, lda, k, laneH);
    v16bf bf = load_b_frag(W, n0 + laneN, Kdim, k, laneH);
    acc0 = wmma_bf16(af0, bf, acc0);
    acc1 = wmma_bf16(af1, bf, acc1);
  }
  const int n = n0 + laneN;
  const float bv = bias[n];
#pragma unroll
  for (int mt = 0; mt < 2; ++mt) {
    const int mb = m0 + mt * 16 + laneH * 8;
#pragma unroll
    for (int r = 0; r < 8; ++r) {
      float v = (mt == 0 ? acc0[r] : acc1[r]) + bv;
      int idx = (mb + r) * N + n;
      outF[idx] = v;
      if (outBF) outBF[idx] = f2bf(v);
    }
  }
}

// --------------------------- small helper kernels ---------------------------

__global__ void convert_bf16_kernel(const float* __restrict__ in,
                                    unsigned short* __restrict__ out, int n) {
  int i = blockIdx.x * blockDim.x + threadIdx.x;
  if (i < n) out[i] = f2bf(in[i]);
}

// xs[t][b][e] = bf16(emb[labels[b][t]][e])
__global__ void embed_kernel(const int* __restrict__ labels,
                             const float* __restrict__ emb,
                             unsigned short* __restrict__ xs, int total) {
  int i = blockIdx.x * blockDim.x + threadIdx.x;
  if (i >= total) return;
  int e = i % En;
  int tb = i / En;
  int b = tb % Bn;
  int t = tb / Bn;
  int lab = labels[b * Tn + t];
  xs[i] = f2bf(emb[lab * En + e]);
}

// h1=c1=h2=c2=h3=c3 = broadcast(h0)   (reference inits c to h as well)
__global__ void init_state_kernel(const float* __restrict__ h0,
                                  float* c1, float* c2, float* c3,
                                  unsigned short* h1b, unsigned short* h2b,
                                  unsigned short* h3b) {
  int i = blockIdx.x * blockDim.x + threadIdx.x;
  if (i >= Bn * Hn) return;
  float v = h0[i % Hn];
  unsigned short vb = f2bf(v);
  c1[i] = v; c2[i] = v; c3[i] = v;
  h1b[i] = vb; h2b[i] = vb; h3b[i] = vb;
}

// energy[b][s] = q[b,:] . keys[s,b,:]   (one wave per dot, wave32 shuffle)
__global__ __launch_bounds__(256) void energy_kernel(
    const float* __restrict__ q, const float* __restrict__ keys,
    float* __restrict__ energy) {
  int lane = threadIdx.x & 31;
  int wid = threadIdx.x >> 5;
  int idx = blockIdx.x * 8 + wid;
  if (idx >= Bn * Sn) return;
  int b = idx / Sn, s = idx % Sn;
  const float4* qp = (const float4*)(q + b * Kn);
  const float4* kp = (const float4*)(keys + (size_t)(s * Bn + b) * Kn);
  float sum = 0.f;
  for (int kk = lane; kk < Kn / 4; kk += 32) {
    float4 a = qp[kk], c = kp[kk];
    sum += a.x * c.x + a.y * c.y + a.z * c.z + a.w * c.w;
  }
  for (int off = 16; off > 0; off >>= 1) sum += __shfl_xor(sum, off, 32);
  if (lane == 0) energy[idx] = sum;
}

// softmax over S per batch row; writes attention straight to d_out slice
__global__ __launch_bounds__(512) void softmax_kernel(
    const float* __restrict__ energy, float* __restrict__ attn_out, int t) {
  __shared__ float red[512];
  int b = blockIdx.x, s = threadIdx.x;
  float e = (s < Sn) ? energy[b * Sn + s] : -3.0e38f;
  red[s] = e;
  __syncthreads();
  for (int off = 256; off > 0; off >>= 1) {
    if (s < off) red[s] = fmaxf(red[s], red[s + off]);
    __syncthreads();
  }
  float m = red[0];
  __syncthreads();
  float p = (s < Sn) ? __expf(e - m) : 0.f;
  red[s] = p;
  __syncthreads();
  for (int off = 256; off > 0; off >>= 1) {
    if (s < off) red[s] += red[s + off];
    __syncthreads();
  }
  float inv = 1.f / red[0];
  if (s < Sn) attn_out[(size_t)b * Tn * Sn + (size_t)t * Sn + s] = p * inv;
}

// ctx[b][k] = sum_s attn[b][s] * values[s][b][k]
__global__ __launch_bounds__(512) void ctx_kernel(
    const float* __restrict__ attn, const float* __restrict__ values,
    float* __restrict__ ctx, unsigned short* __restrict__ ctxb, int t) {
  __shared__ float sa[Sn];
  int b = blockIdx.x, k = threadIdx.x;
  if (k < Sn) sa[k] = attn[(size_t)b * Tn * Sn + (size_t)t * Sn + k];
  __syncthreads();
  float acc = 0.f;
  for (int s = 0; s < Sn; ++s) acc += sa[s] * values[(size_t)(s * Bn + b) * Kn + k];
  ctx[b * Kn + k] = acc;
  ctxb[b * Kn + k] = f2bf(acc);
}

// score[b][v] = [q|ctx] . w_score[v,:] + b_score[v]  (one wave per output)
__global__ __launch_bounds__(256) void score_kernel(
    const float* __restrict__ q, const float* __restrict__ ctx,
    const float* __restrict__ w_score, const float* __restrict__ b_score,
    float* __restrict__ out, int t) {
  int lane = threadIdx.x & 31;
  int wid = threadIdx.x >> 5;
  int idx = blockIdx.x * 8 + wid;
  if (idx >= Bn * Vn) return;
  int b = idx / Vn, v = idx % Vn;
  const float4* wp = (const float4*)(w_score + (size_t)v * (2 * Kn));
  const float4* qp = (const float4*)(q + b * Kn);
  const float4* cp = (const float4*)(ctx + b * Kn);
  float sum = 0.f;
  for (int kk = lane; kk < (2 * Kn) / 4; kk += 32) {
    float4 w4 = wp[kk];
    float4 x4 = (kk < Kn / 4) ? qp[kk] : cp[kk - Kn / 4];
    sum += w4.x * x4.x + w4.y * x4.y + w4.z * x4.z + w4.w * x4.w;
  }
  for (int off = 16; off > 0; off >>= 1) sum += __shfl_xor(sum, off, 32);
  if (lane == 0) out[(size_t)b * Tn * Vn + (size_t)t * Vn + v] = sum + b_score[v];
}

// ---------------------------------------------------------------------------

extern "C" void kernel_launch(void* const* d_in, const int* in_sizes, int n_in,
                              void* d_out, int out_size, void* d_ws, size_t ws_size,
                              hipStream_t stream) {
  (void)in_sizes; (void)n_in; (void)out_size; (void)ws_size;
  const int* labels   = (const int*)d_in[0];
  const float* keys   = (const float*)d_in[1];
  const float* values = (const float*)d_in[2];
  const float* emb    = (const float*)d_in[3];
  const float* w_ih[3] = {(const float*)d_in[4], (const float*)d_in[8],  (const float*)d_in[12]};
  const float* w_hh[3] = {(const float*)d_in[5], (const float*)d_in[9],  (const float*)d_in[13]};
  const float* b_ih[3] = {(const float*)d_in[6], (const float*)d_in[10], (const float*)d_in[14]};
  const float* b_hh[3] = {(const float*)d_in[7], (const float*)d_in[11], (const float*)d_in[15]};
  const float* w_proj  = (const float*)d_in[16];
  const float* b_proj  = (const float*)d_in[17];
  const float* w_score = (const float*)d_in[18];
  const float* b_score = (const float*)d_in[19];
  const float* h0      = (const float*)d_in[20];

  float* out_pred = (float*)d_out;                               // [B,T,V]
  float* out_attn = (float*)d_out + (size_t)Bn * Tn * Vn;        // [B,T,S]

  // ---- workspace carve-out (~61 MB; 256B aligned) ----
  char* ws = (char*)d_ws;
  size_t off = 0;
  auto alloc = [&](size_t bytes) -> void* {
    off = (off + 255) & ~(size_t)255;
    void* p = ws + off;
    off += bytes;
    return p;
  };
  unsigned short* xsb = (unsigned short*)alloc((size_t)Tn * Bn * En * 2);
  unsigned short* wihb[3], *whhb[3];
  for (int c = 0; c < 3; ++c) {
    wihb[c] = (unsigned short*)alloc((size_t)4 * Hn * Hn * 2);  // [4H,1024]
    whhb[c] = (unsigned short*)alloc((size_t)4 * Hn * Hn * 2);  // [4H,1024]
  }
  unsigned short* wpb = (unsigned short*)alloc((size_t)Kn * Hn * 2);
  float* cst[3];
  for (int c = 0; c < 3; ++c) cst[c] = (float*)alloc((size_t)Bn * Hn * 4);
  unsigned short* hb[3][2];
  for (int c = 0; c < 3; ++c)
    for (int p = 0; p < 2; ++p) hb[c][p] = (unsigned short*)alloc((size_t)Bn * Hn * 2);
  float* ctx = (float*)alloc((size_t)Bn * Kn * 4);
  unsigned short* ctxb = (unsigned short*)alloc((size_t)Bn * Kn * 2);
  float* q = (float*)alloc((size_t)Bn * Kn * 4);
  float* energy = (float*)alloc((size_t)Bn * Sn * 4);

  // ---- one-time pre-pass: bf16 weight conversion, embedding, state init ----
  const int WN = 4 * Hn * Hn;  // 4096*1024
  for (int c = 0; c < 3; ++c) {
    convert_bf16_kernel<<<(WN + 255) / 256, 256, 0, stream>>>(w_ih[c], wihb[c], WN);
    convert_bf16_kernel<<<(WN + 255) / 256, 256, 0, stream>>>(w_hh[c], whhb[c], WN);
  }
  convert_bf16_kernel<<<(Kn * Hn + 255) / 256, 256, 0, stream>>>(w_proj, wpb, Kn * Hn);

  const int XT = Tn * Bn * En;
  embed_kernel<<<(XT + 255) / 256, 256, 0, stream>>>(labels, emb, xsb, XT);
  init_state_kernel<<<(Bn * Hn + 255) / 256, 256, 0, stream>>>(
      h0, cst[0], cst[1], cst[2], hb[0][0], hb[1][0], hb[2][0]);

  // ctx0 = h_init @ w_proj^T + b_proj
  gemm_bias_kernel<<<Kn / 32, 128, 0, stream>>>(hb[2][0], Hn, Hn, wpb, b_proj,
                                                ctx, ctxb, Kn);

  // ---- sequential decode loop ----
  for (int t = 0; t < Tn; ++t) {
    int pi = t & 1, ci = pi ^ 1;  // ping-pong h buffers (avoid RAW race in cell GEMM)
    const unsigned short* xt = xsb + (size_t)t * Bn * En;

    // cell 0: input = [x_t (E) | ctx (K)]
    lstm_cell_kernel<<<Hn / 32, 128, 0, stream>>>(
        xt, En, En, ctxb, Kn, Kn, hb[0][pi], wihb[0], Hn, whhb[0],
        b_ih[0], b_hh[0], cst[0], hb[0][ci]);
    // cell 1: input = h1
    lstm_cell_kernel<<<Hn / 32, 128, 0, stream>>>(
        hb[0][ci], Hn, Hn, nullptr, 0, 0, hb[1][pi], wihb[1], Hn, whhb[1],
        b_ih[1], b_hh[1], cst[1], hb[1][ci]);
    // cell 2: input = h2
    lstm_cell_kernel<<<Hn / 32, 128, 0, stream>>>(
        hb[1][ci], Hn, Hn, nullptr, 0, 0, hb[2][pi], wihb[2], Hn, whhb[2],
        b_ih[2], b_hh[2], cst[2], hb[2][ci]);

    // q = h3 @ w_proj^T + b_proj
    gemm_bias_kernel<<<Kn / 32, 128, 0, stream>>>(hb[2][ci], Hn, Hn, wpb, b_proj,
                                                  q, nullptr, Kn);
    // attention
    energy_kernel<<<(Bn * Sn) / 8, 256, 0, stream>>>(q, keys, energy);
    softmax_kernel<<<Bn, 512, 0, stream>>>(energy, out_attn, t);
    ctx_kernel<<<Bn, 512, 0, stream>>>(out_attn, values, ctx, ctxb, t);
    // score = [q|ctx] @ w_score^T + b_score
    score_kernel<<<(Bn * Vn + 7) / 8, 256, 0, stream>>>(q, ctx, w_score, b_score,
                                                        out_pred, t);
  }
}
